// SwinTransformerBlock_18708877541685
// MI455X (gfx1250) — matrix operand order repaired and checked by reading, not compiled, not run on hardware
//
#include <hip/hip_runtime.h>
#include <hip/hip_bf16.h>
#include <cstdint>

// ---------------------------------------------------------------------------
// SwinTransformerBlock for MI455X (gfx1250): bf16 WMMA everywhere it matters.
// B=16, H=W=64, C=256, NH=8, WS=8, HID=1024. Tokens M = 65536.
// ---------------------------------------------------------------------------

typedef __bf16 bf16_t;
typedef __attribute__((ext_vector_type(16))) __bf16 v16bf;
typedef __attribute__((ext_vector_type(8)))  __bf16 v8bf;
typedef __attribute__((ext_vector_type(8)))  float  v8f;

#define DEVINL __device__ __forceinline__

DEVINL bf16_t f2bf(float f) {
    union { float f; uint32_t u; } c; c.f = f;
    uint32_t u = c.u;
    u += 0x7FFFu + ((u >> 16) & 1u);           // round-to-nearest-even
    union { uint16_t s; bf16_t b; } r; r.s = (uint16_t)(u >> 16);
    return r.b;
}
DEVINL float bf2f(bf16_t b) {
    union { uint16_t s; bf16_t b; } c; c.b = b;
    union { uint32_t u; float f; } r; r.u = ((uint32_t)c.s) << 16;
    return r.f;
}

DEVINL v8f wmma_bf16(v16bf a, v16bf b, v8f c) {
    // D = A(16x32) * B(32x16) + C, fp32 accum
    return __builtin_amdgcn_wmma_f32_16x16x32_bf16(
        /*neg_a=*/false, a, /*neg_b=*/false, b,
        /*c_mod=*/(short)0, c, /*reuse_a=*/false, /*reuse_b=*/false);
}

// A fragment (16x32, 16-bit): lane L -> M=L%16, half h=L/16;
// element e -> K = (e%8) + 8h + 16*(e/8)  => two contiguous 8-elem runs.
DEVINL v16bf load_a_frag(const bf16_t* s, int stride, int row, int hl) {
    const v8bf lo = *(const v8bf*)(s + row * stride + 8 * hl);
    const v8bf hi = *(const v8bf*)(s + row * stride + 16 + 8 * hl);
    v16bf a;
#pragma unroll
    for (int i = 0; i < 8; ++i) { a[i] = lo[i]; a[i + 8] = hi[i]; }
    return a;
}
// B fragment (32x16, 16-bit) from B^T stored row-major [n][k]:
// lane L -> N=L%16, half h=L/16; element e -> K = (e%8) + 8h + 16*(e/8).
DEVINL v16bf load_b_frag(const bf16_t* s, int stride, int row, int hl) {
    const v8bf lo = *(const v8bf*)(s + row * stride + 8 * hl);
    const v8bf hi = *(const v8bf*)(s + row * stride + 16 + 8 * hl);
    v16bf b;
#pragma unroll
    for (int i = 0; i < 8; ++i) { b[i] = lo[i]; b[i + 8] = hi[i]; }
    return b;
}

// ---------------------------------------------------------------------------
// fp32 -> bf16 weight conversion
// ---------------------------------------------------------------------------
__global__ void k_f32_to_bf16(const float* __restrict__ src,
                              bf16_t* __restrict__ dst, int n) {
    int i = blockIdx.x * blockDim.x + threadIdx.x;
    if (i < n) dst[i] = f2bf(src[i]);
}

// ---------------------------------------------------------------------------
// LayerNorm over C=256, one wave32 per token, 8 tokens per 256-thread block.
// PERMUTE=true writes rows in window-partition order t' = (b*64+n)*64+s.
// ---------------------------------------------------------------------------
template <bool PERMUTE>
__global__ void k_layernorm(const float* __restrict__ x,
                            const float* __restrict__ w,
                            const float* __restrict__ b,
                            bf16_t* __restrict__ out) {
    const int lane = threadIdx.x & 31;
    const int wv   = threadIdx.x >> 5;
    const int tok  = blockIdx.x * 8 + wv;          // natural token 0..65535
    const float* xr = x + (size_t)tok * 256;
    float v[8];
    float s = 0.f;
#pragma unroll
    for (int i = 0; i < 8; ++i) { v[i] = xr[lane + 32 * i]; s += v[i]; }
#pragma unroll
    for (int m = 1; m < 32; m <<= 1) s += __shfl_xor(s, m, 32);
    float mean = s * (1.f / 256.f);
    float q = 0.f;
#pragma unroll
    for (int i = 0; i < 8; ++i) { float d = v[i] - mean; q += d * d; }
#pragma unroll
    for (int m = 1; m < 32; m <<= 1) q += __shfl_xor(q, m, 32);
    float rstd = rsqrtf(q * (1.f / 256.f) + 1e-5f);

    size_t orow;
    if (PERMUTE) {
        int bI = tok >> 12, p = tok & 4095, hh = p >> 6, ww = p & 63;
        int sW = ((hh >> 3) << 3) + (ww >> 3);
        int nW = ((hh & 7) << 3) + (ww & 7);
        orow = (size_t)(((bI * 64 + nW) << 6) + sW);
    } else {
        orow = (size_t)tok;
    }
    bf16_t* o = out + orow * 256;
#pragma unroll
    for (int i = 0; i < 8; ++i) {
        int c = lane + 32 * i;
        o[c] = f2bf((v[i] - mean) * rstd * w[c] + b[c]);
    }
}

// ---------------------------------------------------------------------------
// Generic bf16 GEMM:  out[M x N] = A[M x K] * W[N x K]^T + bias, LDS-tiled.
// Block: 256 thr (8 waves), tile 128(M) x 64(N), K-step 32.
// Wave (4x2 grid) owns 32x32 = 2x2 WMMA tiles.
// ---------------------------------------------------------------------------
enum { EPI_BF16 = 0, EPI_PROJ = 1, EPI_FC2 = 2 };

template <int EPI>
__global__ void k_gemm(const bf16_t* __restrict__ A,
                       const bf16_t* __restrict__ Wt,
                       const float* __restrict__ bias,
                       const float* __restrict__ resid,
                       bf16_t* __restrict__ outb,
                       float* __restrict__ outf,
                       int K, int Ntot) {
    __shared__ bf16_t sA[128 * 32];
    __shared__ bf16_t sB[64 * 32];
    const int t    = threadIdx.x;
    const int lane = t & 31, wv = t >> 5;
    const int hl   = lane >> 4, l16 = lane & 15;
    const int wm   = (wv >> 1) * 32, wn = (wv & 1) * 32;
    const size_t bm = (size_t)blockIdx.x * 128;
    const int bn   = blockIdx.y * 64;

    v8f acc[2][2] = {};
    const int arow = t >> 1, acol = (t & 1) * 16;   // 128x32 stage: 16 elem/thr
    const int brow = t >> 2, bcol = (t & 3) * 8;    // 64x32 stage: 8 elem/thr

    for (int k0 = 0; k0 < K; k0 += 32) {
        {
            const float4* g = (const float4*)(A + (bm + arow) * (size_t)K + k0 + acol);
            float4* s = (float4*)(sA + arow * 32 + acol);
            s[0] = g[0]; s[1] = g[1];
        }
        {
            const float4* g = (const float4*)(Wt + (size_t)(bn + brow) * K + k0 + bcol);
            *(float4*)(sB + brow * 32 + bcol) = *g;
        }
        __syncthreads();
        v16bf a0 = load_a_frag(sA, 32, wm + l16, hl);
        v16bf a1 = load_a_frag(sA, 32, wm + 16 + l16, hl);
        v16bf b0 = load_b_frag(sB, 32, wn + l16, hl);
        v16bf b1 = load_b_frag(sB, 32, wn + 16 + l16, hl);
        acc[0][0] = wmma_bf16(a0, b0, acc[0][0]);
        acc[0][1] = wmma_bf16(a0, b1, acc[0][1]);
        acc[1][0] = wmma_bf16(a1, b0, acc[1][0]);
        acc[1][1] = wmma_bf16(a1, b1, acc[1][1]);
        __syncthreads();
    }

    // Epilogue. C/D layout: VGPR j -> M = j + 8*hl ; lane%16 -> N column.
#pragma unroll
    for (int mi = 0; mi < 2; ++mi)
#pragma unroll
        for (int ni = 0; ni < 2; ++ni)
#pragma unroll
            for (int j = 0; j < 8; ++j) {
                size_t r = bm + wm + mi * 16 + j + 8 * hl;
                int    c = bn + wn + ni * 16 + l16;
                float  val = acc[mi][ni][j] + bias[c];
                if (EPI == EPI_BF16) {
                    outb[r * (size_t)Ntot + c] = f2bf(val);
                } else if (EPI == EPI_PROJ) {
                    // rows are t' = (b*64+n)*64+s; window-reverse + residual
                    int bI = (int)(r >> 12), nW = (int)((r >> 6) & 63), sW = (int)(r & 63);
                    int hh = ((sW >> 3) << 3) + (nW >> 3);
                    int ww = ((sW & 7) << 3) + (nW & 7);
                    size_t nat = (size_t)bI * 4096 + hh * 64 + ww;
                    outf[nat * 256 + c] = val + resid[nat * 256 + c];
                } else { // EPI_FC2: natural rows, residual from x1 (= d_out)
                    outf[r * 256 + c] = val + resid[r * 256 + c];
                }
            }
}

// ---------------------------------------------------------------------------
// Attention logits + rel-bias + softmax. One block per (b,h): 64x64 over
// d=2048. qkv rows are t'=(b*64+n)*64+s so Q/K/V slices are contiguous,
// row stride 49152 bf16. Block: 128 thr (4 waves), wave = 16 rows x 64 cols.
// ---------------------------------------------------------------------------
__global__ void k_attn_softmax(const bf16_t* __restrict__ qkv,
                               const float* __restrict__ rel_bias,
                               bf16_t* __restrict__ probs) {
    __shared__ bf16_t sQ[64 * 32];
    __shared__ bf16_t sK[64 * 32];
    const int bh = blockIdx.x;                 // b*8 + h
    const int bI = bh >> 3, hI = bh & 7;
    const int t = threadIdx.x;
    const int lane = t & 31, wv = t >> 5;
    const int hl = lane >> 4, l16 = lane & 15;
    const size_t base = (size_t)bI * 64 * 49152 + (size_t)hI * 2048;

    v8f acc[4] = {};
    const int row = t >> 1, colo = (t & 1) * 16;

    for (int k0 = 0; k0 < 2048; k0 += 32) {
        {
            const float4* gq = (const float4*)(qkv + base + (size_t)row * 49152 + k0 + colo);
            float4* sq = (float4*)(sQ + row * 32 + colo);
            sq[0] = gq[0]; sq[1] = gq[1];
            const float4* gk = (const float4*)(qkv + base + 16384 + (size_t)row * 49152 + k0 + colo);
            float4* sk = (float4*)(sK + row * 32 + colo);
            sk[0] = gk[0]; sk[1] = gk[1];
        }
        __syncthreads();
        v16bf a = load_a_frag(sQ, 32, wv * 16 + l16, hl);
#pragma unroll
        for (int ti = 0; ti < 4; ++ti) {
            v16bf bfr = load_b_frag(sK, 32, ti * 16 + l16, hl);
            acc[ti] = wmma_bf16(a, bfr, acc[ti]);
        }
        __syncthreads();
    }

    const float scale = 0.17677669529663687f;   // (C/NH)^-0.5 = 1/sqrt(32)
    float lg[8][4];
#pragma unroll
    for (int j = 0; j < 8; ++j) {
        int n = wv * 16 + j + 8 * hl;
        int nr = n >> 3, nc = n & 7;
#pragma unroll
        for (int ti = 0; ti < 4; ++ti) {
            int m = ti * 16 + l16;
            int mr = m >> 3, mc = m & 7;
            int idx = (nr - mr + 7) * 15 + (nc - mc + 7);   // rel pos index
            lg[j][ti] = acc[ti][j] * scale + rel_bias[idx * 8 + hI];
        }
    }
#pragma unroll
    for (int j = 0; j < 8; ++j) {
        float mx = lg[j][0];
#pragma unroll
        for (int ti = 1; ti < 4; ++ti) mx = fmaxf(mx, lg[j][ti]);
#pragma unroll
        for (int msk = 1; msk < 16; msk <<= 1) mx = fmaxf(mx, __shfl_xor(mx, msk, 32));
        float sum = 0.f;
#pragma unroll
        for (int ti = 0; ti < 4; ++ti) { lg[j][ti] = __expf(lg[j][ti] - mx); sum += lg[j][ti]; }
#pragma unroll
        for (int msk = 1; msk < 16; msk <<= 1) sum += __shfl_xor(sum, msk, 32);
        float inv = 1.f / sum;
        int n = wv * 16 + j + 8 * hl;
        bf16_t* orow = probs + (size_t)bh * 4096 + n * 64;
#pragma unroll
        for (int ti = 0; ti < 4; ++ti) orow[ti * 16 + l16] = f2bf(lg[j][ti] * inv);
    }
}

// ---------------------------------------------------------------------------
// out = P(64x64) @ V(64x2048) per (b,h). Grid (b*h, d-tile of 64).
// V staged transposed into LDS so B-frags are contiguous reads.
// Output stored as ao[(b*64+n)*16384 + h*2048 + d] (== t'-order tokens x C).
// ---------------------------------------------------------------------------
__global__ void k_attn_v(const bf16_t* __restrict__ qkv,
                         const bf16_t* __restrict__ probs,
                         bf16_t* __restrict__ ao) {
    __shared__ bf16_t sVt[64 * 32];            // [d_local][m_local]
    const int bh = blockIdx.x;
    const int dt = blockIdx.y;                 // 0..31, d0 = dt*64
    const int bI = bh >> 3, hI = bh & 7;
    const int t = threadIdx.x;                 // 128 threads, 4 waves
    const int lane = t & 31, wv = t >> 5;
    const int hl = lane >> 4, l16 = lane & 15;
    const size_t vbase = (size_t)bI * 64 * 49152 + 32768 + (size_t)hI * 2048 + dt * 64;

    v8f acc[4] = {};
    const int m_ld = t >> 2, doff = (t & 3) * 16;

    for (int k0 = 0; k0 < 64; k0 += 32) {
        {   // stage V[m][d] transposed -> sVt[d][m]
            const bf16_t* vp = qkv + vbase + (size_t)(k0 + m_ld) * 49152 + doff;
            v8bf v0 = *(const v8bf*)(vp);
            v8bf v1 = *(const v8bf*)(vp + 8);
#pragma unroll
            for (int i = 0; i < 8; ++i) {
                sVt[(doff + i) * 32 + m_ld]     = v0[i];
                sVt[(doff + 8 + i) * 32 + m_ld] = v1[i];
            }
        }
        __syncthreads();
        // A-frag straight from global (P rows are contiguous 64 bf16)
        const bf16_t* prow = probs + (size_t)bh * 4096 + (wv * 16 + l16) * 64 + k0;
        v8bf lo = *(const v8bf*)(prow + 8 * hl);
        v8bf hi = *(const v8bf*)(prow + 16 + 8 * hl);
        v16bf a;
#pragma unroll
        for (int i = 0; i < 8; ++i) { a[i] = lo[i]; a[i + 8] = hi[i]; }
#pragma unroll
        for (int ti = 0; ti < 4; ++ti) {
            v16bf bfr = load_b_frag(sVt, 32, ti * 16 + l16, hl);
            acc[ti] = wmma_bf16(a, bfr, acc[ti]);
        }
        __syncthreads();
    }
#pragma unroll
    for (int ti = 0; ti < 4; ++ti)
#pragma unroll
        for (int j = 0; j < 8; ++j) {
            int n = wv * 16 + j + 8 * hl;
            int d = dt * 64 + ti * 16 + l16;
            ao[((size_t)bI * 64 + n) * 16384 + (size_t)hI * 2048 + d] = f2bf(acc[ti][j]);
        }
}

// ---------------------------------------------------------------------------
// Depthwise 3x3 (SAME, zero pad) + bias + exact GELU. 4 channels/thread.
// ---------------------------------------------------------------------------
__global__ void k_dwconv_gelu(const bf16_t* __restrict__ h1,
                              const float* __restrict__ dw_w,
                              const float* __restrict__ dw_b,
                              bf16_t* __restrict__ h2) {
    size_t gid = (size_t)blockIdx.x * blockDim.x + threadIdx.x;
    int c0 = (int)(gid & 255) * 4;
    size_t pix = gid >> 8;                     // 0..65535
    int bI = (int)(pix >> 12), p = (int)(pix & 4095);
    int hh = p >> 6, ww = p & 63;
    float accv[4] = {0.f, 0.f, 0.f, 0.f};
#pragma unroll
    for (int ky = -1; ky <= 1; ++ky) {
        int yy = hh + ky; if (yy < 0 || yy > 63) continue;
#pragma unroll
        for (int kx = -1; kx <= 1; ++kx) {
            int xx = ww + kx; if (xx < 0 || xx > 63) continue;
            const bf16_t* src = h1 + ((size_t)bI * 4096 + yy * 64 + xx) * 1024 + c0;
            const float* wv4 = dw_w + ((ky + 1) * 3 + (kx + 1)) * 1024 + c0;
#pragma unroll
            for (int i = 0; i < 4; ++i) accv[i] += bf2f(src[i]) * wv4[i];
        }
    }
    bf16_t* dst = h2 + pix * 1024 + c0;
#pragma unroll
    for (int i = 0; i < 4; ++i) {
        float v = accv[i] + dw_b[c0 + i];
        float g = 0.5f * v * (1.f + erff(v * 0.70710678118654752f));
        dst[i] = f2bf(g);
    }
}

// ---------------------------------------------------------------------------
// Launch. Workspace layout (bytes):
//   [0)               bf16 weights: qkv_w 196608 | proj_w 65536 | fc1_w 262144 | fc2_w 262144
//   [1,572,864)       wn   : 65536x256  bf16 (LN1 out, reused for LN2 out)
//   [35,127,296)      wqkv : 65536x768  bf16   (dead after attn -> reused by wh1)
//   [135,790,592)     wprob: 128x64x64  bf16
//   [136,839,168)     wao  : 65536x256  bf16
//   wh1 = 35,127,296  : 65536x1024 bf16   (overlaps wqkv/wprob/wao, all dead)
//   wh2 = 169,345,024 : 65536x1024 bf16   -> total ~304 MB
// ---------------------------------------------------------------------------
extern "C" void kernel_launch(void* const* d_in, const int* in_sizes, int n_in,
                              void* d_out, int out_size, void* d_ws, size_t ws_size,
                              hipStream_t stream) {
    (void)in_sizes; (void)n_in; (void)out_size; (void)ws_size;
    const float* x       = (const float*)d_in[0];
    const float* norm1_w = (const float*)d_in[1];
    const float* norm1_b = (const float*)d_in[2];
    const float* qkv_w   = (const float*)d_in[3];
    const float* qkv_b   = (const float*)d_in[4];
    const float* rel_b   = (const float*)d_in[5];
    const float* proj_w  = (const float*)d_in[6];
    const float* proj_b  = (const float*)d_in[7];
    const float* norm2_w = (const float*)d_in[8];
    const float* norm2_b = (const float*)d_in[9];
    const float* fc1_w   = (const float*)d_in[10];
    const float* fc1_b   = (const float*)d_in[11];
    const float* dw_w    = (const float*)d_in[12];
    const float* dw_b    = (const float*)d_in[13];
    const float* fc2_w   = (const float*)d_in[14];
    const float* fc2_b   = (const float*)d_in[15];
    float* out = (float*)d_out;

    char* ws = (char*)d_ws;
    bf16_t* w_qkv = (bf16_t*)(ws + 0);
    bf16_t* w_prj = w_qkv + 768 * 256;
    bf16_t* w_fc1 = w_prj + 256 * 256;
    bf16_t* w_fc2 = w_fc1 + 1024 * 256;
    bf16_t* wn    = (bf16_t*)(ws + 1572864ull);
    bf16_t* wqkv  = (bf16_t*)(ws + 35127296ull);
    bf16_t* wprob = (bf16_t*)(ws + 135790592ull);
    bf16_t* wao   = (bf16_t*)(ws + 136839168ull);
    bf16_t* wh1   = (bf16_t*)(ws + 35127296ull);     // reuse (wqkv dead)
    bf16_t* wh2   = (bf16_t*)(ws + 169345024ull);

    // 0) weights -> bf16
    k_f32_to_bf16<<<(196608 + 255) / 256, 256, 0, stream>>>(qkv_w, w_qkv, 196608);
    k_f32_to_bf16<<<(65536 + 255) / 256, 256, 0, stream>>>(proj_w, w_prj, 65536);
    k_f32_to_bf16<<<(262144 + 255) / 256, 256, 0, stream>>>(fc1_w, w_fc1, 262144);
    k_f32_to_bf16<<<(262144 + 255) / 256, 256, 0, stream>>>(fc2_w, w_fc2, 262144);

    // 1) LN1, rows in window-partition order t'
    k_layernorm<true><<<8192, 256, 0, stream>>>(x, norm1_w, norm1_b, wn);

    // 2) qkv GEMM: 65536x256 @ 768x256^T
    k_gemm<EPI_BF16><<<dim3(512, 12), 256, 0, stream>>>(
        wn, w_qkv, qkv_b, nullptr, wqkv, nullptr, 256, 768);

    // 3) attention logits + softmax (per b,h; d=2048)
    k_attn_softmax<<<128, 128, 0, stream>>>(wqkv, rel_b, wprob);

    // 4) P @ V
    k_attn_v<<<dim3(128, 32), 128, 0, stream>>>(wqkv, wprob, wao);

    // 5) proj GEMM + window reverse + residual -> d_out (x1, fp32)
    k_gemm<EPI_PROJ><<<dim3(512, 4), 256, 0, stream>>>(
        wao, w_prj, proj_b, x, nullptr, out, 256, 256);

    // 6) LN2 (natural order) from x1
    k_layernorm<false><<<8192, 256, 0, stream>>>(out, norm2_w, norm2_b, wn);

    // 7) fc1 GEMM: 65536x256 @ 1024x256^T
    k_gemm<EPI_BF16><<<dim3(512, 16), 256, 0, stream>>>(
        wn, w_fc1, fc1_b, nullptr, wh1, nullptr, 256, 1024);

    // 8) depthwise 3x3 + GELU
    k_dwconv_gelu<<<65536, 256, 0, stream>>>(wh1, dw_w, dw_b, wh2);

    // 9) fc2 GEMM + residual -> d_out
    k_gemm<EPI_FC2><<<dim3(512, 4), 256, 0, stream>>>(
        wh2, w_fc2, fc2_b, out, nullptr, out, 1024, 256);
}
